// LstmDecoder_67894843015841
// MI455X (gfx1250) — matrix operand order repaired and checked by reading, not compiled, not run on hardware
//
#include <hip/hip_runtime.h>
#include <hip/hip_bf16.h>
#include <stdint.h>
#include <stddef.h>

// ---------------- problem constants ----------------
#define BATCH 32
#define TLEN  128
#define HDIM  1024
#define ZDIM  512
#define ODIM  512
#define IDIM  1024        // OUT + Z
#define G4    4096        // 4*HDIM

typedef __attribute__((ext_vector_type(16))) __bf16 v16bf;
typedef __attribute__((ext_vector_type(8)))  __bf16 v8bf;
typedef __attribute__((ext_vector_type(8)))  float  v8f;
typedef __attribute__((ext_vector_type(4)))  int    v4i;

// Async global->LDS path if the toolchain declares the gfx1250 builtins.
#if defined(__HIP_DEVICE_COMPILE__) && \
    __has_builtin(__builtin_amdgcn_global_load_async_to_lds_b128) && \
    __has_builtin(__builtin_amdgcn_s_wait_asynccnt)
#define USE_ASYNC 1
#else
#define USE_ASYNC 0
#endif

// ---------------- helpers ----------------
__device__ __forceinline__ __bf16 f2bf(float f) {
    union { float f; uint32_t u; } a; a.f = f;
    uint32_t u = a.u;
    uint32_t r = (u + 0x7FFFu + ((u >> 16) & 1u)) >> 16;
    union { unsigned short s; __bf16 b; } o; o.s = (unsigned short)r;
    return o.b;
}

__device__ __forceinline__ float sigmoidf_(float x) {
    return 1.0f / (1.0f + __expf(-x));
}

#if USE_ASYNC
// generic -> as(1)/as(3) v4i* via integer casts (LDS offset = low 32 bits of flat addr)
__device__ __forceinline__ __attribute__((address_space(1))) v4i* as1v(const void* p) {
    return (__attribute__((address_space(1))) v4i*)(unsigned long long)p;
}
__device__ __forceinline__ __attribute__((address_space(3))) v4i* as3v(void* p) {
    return (__attribute__((address_space(3))) v4i*)(unsigned long long)(unsigned)(unsigned long long)p;
}
#endif

// A-fragment: 16x32 bf16, row-major source with leading dim K.
// lane l: row = m0 + (l&15); half = l>>4.
// elements 0..7  = K k0 + half*8 + j ; elements 8..15 = K k0 + 16 + half*8 + j
__device__ __forceinline__ v16bf load_afrag(const __bf16* __restrict__ base,
                                            int m0, int K, int k0, int lane) {
    int row  = m0 + (lane & 15);
    int half = lane >> 4;
    const __bf16* p = base + (size_t)row * K + k0 + half * 8;
    v8bf lo = *(const v8bf*)(p);
    v8bf hi = *(const v8bf*)(p + 16);
    v16bf a;
#pragma unroll
    for (int i = 0; i < 8; ++i) { a[i] = lo[i]; a[i + 8] = hi[i]; }
    return a;
}

// B-fragment from W row-major [N,K] (B[k][n] = W[n][k]).
// lane l: col = n0 + (l&15); half = l>>4; elements 0..7 = K k0 + half*8 + j ; elements 8..15 = K k0 + 16 + half*8 + j
__device__ __forceinline__ v16bf load_bfrag(const __bf16* __restrict__ Wrow,
                                            int n0, int K, int k0, int lane) {
    int col  = n0 + (lane & 15);
    int half = lane >> 4;
    const __bf16* p = Wrow + (size_t)col * K + k0 + half * 8;
    v8bf lo = *(const v8bf*)(p);
    v8bf hi = *(const v8bf*)(p + 16);
    v16bf b;
#pragma unroll
    for (int i = 0; i < 8; ++i) { b[i] = lo[i]; b[i + 8] = hi[i]; }
    return b;
}

// B-fragment out of an LDS tile with padded row stride LDSTR (elements).
#define LDSTR 40   // 32 K-elements + 8 pad -> 80B row stride, bank-conflict-free b128 reads
__device__ __forceinline__ v16bf lds_bfrag(const __bf16* __restrict__ tile,
                                           int nsub, int lane) {
    int col  = (nsub << 4) + (lane & 15);
    int half = lane >> 4;
    const __bf16* p = tile + col * LDSTR + half * 8;
    v8bf lo = *(const v8bf*)(p);
    v8bf hi = *(const v8bf*)(p + 16);
    v16bf b;
#pragma unroll
    for (int i = 0; i < 8; ++i) { b[i] = lo[i]; b[i + 8] = hi[i]; }
    return b;
}

// ---------------- kernel: f32 -> bf16 convert ----------------
__global__ void k_f32_to_bf16(const float* __restrict__ in, __bf16* __restrict__ out, int n) {
    int i = blockIdx.x * blockDim.x + threadIdx.x;
    if (i < n) out[i] = f2bf(in[i]);
}

// ---------------- kernel: build x = concat(seq, z) in [t*B+b, 1024] bf16 ----------------
__global__ void k_build_x(const float* __restrict__ seq,   // [B, T, ODIM]
                          const float* __restrict__ z,     // [B, ZDIM]
                          __bf16* __restrict__ x) {        // [T*B, IDIM]
    int i = blockIdx.x * blockDim.x + threadIdx.x;
    int j   = i & (IDIM - 1);
    int row = i >> 10;               // t*B + b
    int b   = row & (BATCH - 1);
    int t   = row >> 5;
    float v = (j < ODIM) ? seq[((size_t)b * TLEN + t) * ODIM + j]
                         : z[(size_t)b * ZDIM + (j - ODIM)];
    x[i] = f2bf(v);
}

// ---------------- kernel: initial states tanh(z @ W_embed^T + b) ----------------
__global__ void k_init_state(const float* __restrict__ z,
                             const float* __restrict__ We,
                             const float* __restrict__ be,
                             __bf16* __restrict__ h0l0, __bf16* __restrict__ h0l1,
                             float*  __restrict__ c0,   float*  __restrict__ c1) {
    int i = blockIdx.x * blockDim.x + threadIdx.x;   // B*4096
    int b = i >> 12;
    int g = i & 4095;
    const float* zr = z  + (size_t)b * ZDIM;
    const float* wr = We + (size_t)g * ZDIM;
    float acc = be[g];
#pragma unroll 4
    for (int k = 0; k < ZDIM; ++k) acc += zr[k] * wr[k];
    float v = tanhf(acc);
    int l = g >> 11, s = (g >> 10) & 1, hidx = g & (HDIM - 1);
    size_t idx = (size_t)b * HDIM + hidx;
    if (s == 0) { if (l == 0) h0l0[idx] = f2bf(v); else h0l1[idx] = f2bf(v); }
    else        { if (l == 0) c0[idx]   = v;       else c1[idx]   = v; }
}

// ---------------- kernel: bf16 WMMA GEMM with LDS-staged B, 64x64 tile per WG --------
// C[M,N] f32 = A[M,K]bf16 @ W[N,K]bf16^T + bias0[n] + bias1[n]
// 128 threads (4 waves); wave w computes rows [wgM+16w, +16) x cols [wgN, +64).
// B tile (64N x 32K) staged per k-step into double-buffered LDS, shared by all waves.
__global__ void k_gemm_bf16(const __bf16* __restrict__ A,
                            const __bf16* __restrict__ W,
                            const float* __restrict__ bias0,
                            const float* __restrict__ bias1,
                            float* __restrict__ C,
                            int M, int N, int K, int obatch, int oT) {
    __shared__ __bf16 sB[2][64 * LDSTR];

    const int tid  = threadIdx.x;
    const int lane = tid & 31;
    const int wv   = tid >> 5;
    const int nblk = N >> 6;
    const int wgN  = (blockIdx.x % nblk) << 6;
    const int wgM  = (blockIdx.x / nblk) << 6;
    const int m0   = wgM + (wv << 4);

    // staging: thread stages 16 K-elements (32B) of row srow
    const int srow  = tid >> 1;
    const int skoff = (tid & 1) << 4;
    const __bf16* gB = W + (size_t)(wgN + srow) * K + skoff;
    __bf16* sdst[2] = { &sB[0][srow * LDSTR + skoff], &sB[1][srow * LDSTR + skoff] };

    // A stream pointer (per-lane)
    const __bf16* gA = A + (size_t)(m0 + (lane & 15)) * K + ((lane >> 4) << 3);

    v8f acc0 = {}, acc1 = {}, acc2 = {}, acc3 = {};

    auto stage = [&](int buf, int k0) {
#if USE_ASYNC
        __builtin_amdgcn_global_load_async_to_lds_b128(as1v(gB + k0),     as3v(sdst[buf]),     0, 0);
        __builtin_amdgcn_global_load_async_to_lds_b128(as1v(gB + k0 + 8), as3v(sdst[buf] + 8), 0, 0);
#else
        v8bf t0 = *(const v8bf*)(gB + k0);
        v8bf t1 = *(const v8bf*)(gB + k0 + 8);
        *(v8bf*)(sdst[buf])     = t0;
        *(v8bf*)(sdst[buf] + 8) = t1;
#endif
    };

    stage(0, 0);
    const int nk = K >> 5;
    for (int i = 0; i < nk; ++i) {
        const int k0  = i << 5;
        const int buf = i & 1;
#if USE_ASYNC
        __builtin_amdgcn_s_wait_asynccnt(0);
#endif
        __syncthreads();
        if (i + 1 < nk) stage(buf ^ 1, k0 + 32);

        // A fragment (direct global, per-lane b128 pair)
        v8bf alo = *(const v8bf*)(gA + k0);
        v8bf ahi = *(const v8bf*)(gA + k0 + 16);
        v16bf a;
#pragma unroll
        for (int e = 0; e < 8; ++e) { a[e] = alo[e]; a[e + 8] = ahi[e]; }

        v16bf b0 = lds_bfrag(sB[buf], 0, lane);
        v16bf b1 = lds_bfrag(sB[buf], 1, lane);
        v16bf b2 = lds_bfrag(sB[buf], 2, lane);
        v16bf b3 = lds_bfrag(sB[buf], 3, lane);
        acc0 = __builtin_amdgcn_wmma_f32_16x16x32_bf16(false, a, false, b0, (short)0, acc0, false, false);
        acc1 = __builtin_amdgcn_wmma_f32_16x16x32_bf16(false, a, false, b1, (short)0, acc1, false, false);
        acc2 = __builtin_amdgcn_wmma_f32_16x16x32_bf16(false, a, false, b2, (short)0, acc2, false, false);
        acc3 = __builtin_amdgcn_wmma_f32_16x16x32_bf16(false, a, false, b3, (short)0, acc3, false, false);
    }

    const int nl = lane & 15, half = lane >> 4;
#pragma unroll
    for (int j = 0; j < 4; ++j) {
        v8f acc = (j == 0) ? acc0 : (j == 1) ? acc1 : (j == 2) ? acc2 : acc3;
        int ng = wgN + (j << 4) + nl;
        float bias = 0.f;
        if (bias0) bias += bias0[ng];
        if (bias1) bias += bias1[ng];
#pragma unroll
        for (int r = 0; r < 8; ++r) {
            int m = m0 + r + 8 * half;
            size_t row = (obatch > 0) ? ((size_t)(m % obatch) * oT + (m / obatch)) : (size_t)m;
            C[row * N + ng] = acc[r] + bias;
        }
    }
}

// ---------------- kernel: one LSTM time step (recurrent GEMM + elementwise) ----------
// Register double-buffered fragments: WMMA at step k consumes step k's fragments while
// step k+32 loads are in flight. A fragment shared by the 4 gate WMMAs.
__global__ void k_lstm_step(const float*  __restrict__ xp_t,   // [B, 4H] slice for step t
                            const __bf16* __restrict__ hprev,  // [B, H]
                            const __bf16* __restrict__ Whh,    // [4H, H]
                            float*  __restrict__ cbuf,         // [B, H] in/out
                            __bf16* __restrict__ hout) {       // [B, H] = hs[t]
    int wid  = blockIdx.x * (blockDim.x >> 5) + (threadIdx.x >> 5);
    int lane = threadIdx.x & 31;
    int nt = wid & 63;
    int mt = wid >> 6;
    int m0 = mt << 4, n0 = nt << 4;

    v8f ai = {}, af = {}, ag = {}, ao = {};

    v16bf a_c  = load_afrag(hprev, m0, HDIM, 0, lane);
    v16bf bi_c = load_bfrag(Whh, 0 * HDIM + n0, HDIM, 0, lane);
    v16bf bf_c = load_bfrag(Whh, 1 * HDIM + n0, HDIM, 0, lane);
    v16bf bg_c = load_bfrag(Whh, 2 * HDIM + n0, HDIM, 0, lane);
    v16bf bo_c = load_bfrag(Whh, 3 * HDIM + n0, HDIM, 0, lane);

    for (int k0 = 0; k0 < HDIM; k0 += 32) {
        const int kn = k0 + 32;
        v16bf a_n, bi_n, bf_n, bg_n, bo_n;
        if (kn < HDIM) {
            a_n  = load_afrag(hprev, m0, HDIM, kn, lane);
            bi_n = load_bfrag(Whh, 0 * HDIM + n0, HDIM, kn, lane);
            bf_n = load_bfrag(Whh, 1 * HDIM + n0, HDIM, kn, lane);
            bg_n = load_bfrag(Whh, 2 * HDIM + n0, HDIM, kn, lane);
            bo_n = load_bfrag(Whh, 3 * HDIM + n0, HDIM, kn, lane);
        }
        ai = __builtin_amdgcn_wmma_f32_16x16x32_bf16(false, a_c, false, bi_c, (short)0, ai, false, false);
        af = __builtin_amdgcn_wmma_f32_16x16x32_bf16(false, a_c, false, bf_c, (short)0, af, false, false);
        ag = __builtin_amdgcn_wmma_f32_16x16x32_bf16(false, a_c, false, bg_c, (short)0, ag, false, false);
        ao = __builtin_amdgcn_wmma_f32_16x16x32_bf16(false, a_c, false, bo_c, (short)0, ao, false, false);
        if (kn < HDIM) {
            a_c = a_n; bi_c = bi_n; bf_c = bf_n; bg_c = bg_n; bo_c = bo_n;
        }
    }

    int nl = lane & 15, half = lane >> 4;
    int n = n0 + nl;
#pragma unroll
    for (int r = 0; r < 8; ++r) {
        int m = m0 + r + 8 * half;
        const float* xr = xp_t + (size_t)m * G4;
        float gi = ai[r] + xr[0 * HDIM + n];
        float gf = af[r] + xr[1 * HDIM + n];
        float gg = ag[r] + xr[2 * HDIM + n];
        float go = ao[r] + xr[3 * HDIM + n];
        float iv = sigmoidf_(gi);
        float fv = sigmoidf_(gf);
        float gv = tanhf(gg);
        float ov = sigmoidf_(go);
        size_t idx = (size_t)m * HDIM + n;
        float cn = fv * cbuf[idx] + iv * gv;
        cbuf[idx] = cn;
        hout[idx] = f2bf(ov * tanhf(cn));
    }
}

// ---------------- kernel: tail scalar (sampled_frac = 0) ----------------
__global__ void k_tail(float* out, int idx) {
    if (threadIdx.x == 0 && blockIdx.x == 0) out[idx] = 0.0f;
}

// ---------------- host launcher ----------------
extern "C" void kernel_launch(void* const* d_in, const int* in_sizes, int n_in,
                              void* d_out, int out_size, void* d_ws, size_t ws_size,
                              hipStream_t stream) {
    const float* z      = (const float*)d_in[0];
    const float* seq    = (const float*)d_in[1];
    const float* W_emb  = (const float*)d_in[3];
    const float* b_emb  = (const float*)d_in[4];
    const float* W_ih0  = (const float*)d_in[5];
    const float* W_hh0  = (const float*)d_in[6];
    const float* b_ih0  = (const float*)d_in[7];
    const float* b_hh0  = (const float*)d_in[8];
    const float* W_ih1  = (const float*)d_in[9];
    const float* W_hh1  = (const float*)d_in[10];
    const float* b_ih1  = (const float*)d_in[11];
    const float* b_hh1  = (const float*)d_in[12];
    const float* W_out  = (const float*)d_in[13];
    const float* b_out  = (const float*)d_in[14];
    float* out = (float*)d_out;

    char* ws = (char*)d_ws;
    size_t off = 0;
    auto alloc = [&](size_t bytes) -> char* {
        char* p = ws + off;
        off += (bytes + 255) & ~(size_t)255;
        return p;
    };
    __bf16* xbf   = (__bf16*)alloc((size_t)TLEN * BATCH * IDIM * 2);
    __bf16* wih0  = (__bf16*)alloc((size_t)G4 * IDIM * 2);
    __bf16* whh0  = (__bf16*)alloc((size_t)G4 * HDIM * 2);
    __bf16* wih1  = (__bf16*)alloc((size_t)G4 * HDIM * 2);
    __bf16* whh1  = (__bf16*)alloc((size_t)G4 * HDIM * 2);
    __bf16* wout  = (__bf16*)alloc((size_t)ODIM * HDIM * 2);
    __bf16* h0l0  = (__bf16*)alloc((size_t)BATCH * HDIM * 2);
    __bf16* h0l1  = (__bf16*)alloc((size_t)BATCH * HDIM * 2);
    float*  c0    = (float*) alloc((size_t)BATCH * HDIM * 4);
    float*  c1    = (float*) alloc((size_t)BATCH * HDIM * 4);
    __bf16* hs0   = (__bf16*)alloc((size_t)TLEN * BATCH * HDIM * 2);
    __bf16* hs1   = (__bf16*)alloc((size_t)TLEN * BATCH * HDIM * 2);
    float*  xp    = (float*) alloc((size_t)TLEN * BATCH * G4 * 4);
    (void)ws_size; (void)in_sizes; (void)n_in;

    // 1. weight conversions to bf16
    {
        int n = G4 * IDIM;
        k_f32_to_bf16<<<(n + 255) / 256, 256, 0, stream>>>(W_ih0, wih0, n);
        n = G4 * HDIM;
        k_f32_to_bf16<<<(n + 255) / 256, 256, 0, stream>>>(W_hh0, whh0, n);
        k_f32_to_bf16<<<(n + 255) / 256, 256, 0, stream>>>(W_hh1, whh1, n);
        k_f32_to_bf16<<<(n + 255) / 256, 256, 0, stream>>>(W_ih1, wih1, n);
        n = ODIM * HDIM;
        k_f32_to_bf16<<<(n + 255) / 256, 256, 0, stream>>>(W_out, wout, n);
    }

    // 2. build x, initial states
    {
        int n = TLEN * BATCH * IDIM;
        k_build_x<<<n / 256, 256, 0, stream>>>(seq, z, xbf);
        k_init_state<<<(BATCH * G4) / 256, 256, 0, stream>>>(z, W_emb, b_emb,
                                                             h0l0, h0l1, c0, c1);
    }

    const int MROWS = TLEN * BATCH;   // 4096

    // 3. xp0 = x @ W_ih0^T + b_ih0 + b_hh0
    {
        int blocks = (MROWS / 64) * (G4 / 64);
        k_gemm_bf16<<<blocks, 128, 0, stream>>>(xbf, wih0, b_ih0, b_hh0,
                                                xp, MROWS, G4, IDIM, 0, 0);
    }

    // 4. layer-0 sequential scan
    {
        const __bf16* hprev = h0l0;
        for (int t = 0; t < TLEN; ++t) {
            k_lstm_step<<<32, 128, 0, stream>>>(xp + (size_t)t * BATCH * G4, hprev,
                                                whh0, c0,
                                                hs0 + (size_t)t * BATCH * HDIM);
            hprev = hs0 + (size_t)t * BATCH * HDIM;
        }
    }

    // 5. xp1 = hs0 @ W_ih1^T + b_ih1 + b_hh1  (reuse xp buffer)
    {
        int blocks = (MROWS / 64) * (G4 / 64);
        k_gemm_bf16<<<blocks, 128, 0, stream>>>(hs0, wih1, b_ih1, b_hh1,
                                                xp, MROWS, G4, HDIM, 0, 0);
    }

    // 6. layer-1 sequential scan
    {
        const __bf16* hprev = h0l1;
        for (int t = 0; t < TLEN; ++t) {
            k_lstm_step<<<32, 128, 0, stream>>>(xp + (size_t)t * BATCH * G4, hprev,
                                                whh1, c1,
                                                hs1 + (size_t)t * BATCH * HDIM);
            hprev = hs1 + (size_t)t * BATCH * HDIM;
        }
    }

    // 7. out[b,t,:] = hs1 @ W_out^T + b_out  (row swizzle t*B+b -> b*T+t)
    {
        int blocks = (MROWS / 64) * (ODIM / 64);
        k_gemm_bf16<<<blocks, 128, 0, stream>>>(hs1, wout, b_out, nullptr,
                                                out, MROWS, ODIM, HDIM,
                                                BATCH, TLEN);
    }

    // 8. sampled_frac scalar
    k_tail<<<1, 32, 0, stream>>>(out, out_size - 1);
}
